// FlexibleARCEncoder_16466904613258
// MI455X (gfx1250) — matrix operand order repaired and checked by reading, not compiled, not run on hardware
//
#include <hip/hip_runtime.h>

// ---------------------------------------------------------------------------
// FlexibleARCEncoder fused forward for MI455X (gfx1250, wave32, WMMA).
//
// Roofline: ~60 GFLOP, <10 MB mandatory HBM traffic @ 23.3 TB/s -> compute
// bound ONLY if fully fused. All per-pixel intermediates live in LDS/regs;
// only O(B*32*(66+64)) floats of softmax / mean partials hit memory.
// Matrix math: v_wmma_f32_16x16x32_bf16.
//
// v2: weights pre-packed into WMMA B-fragment order (2x ds_load_b128/frag).
// v3: software-pipelined B loads (compiler re-serialized them).
// v4: batch all B fragments of a K-chunk into a register array before the
//     WMMA chain -> independent registers force overlapped partial dscnt
//     waits instead of s_wait_dscnt 0x0 before every wmma.
// ---------------------------------------------------------------------------

typedef __attribute__((ext_vector_type(16))) __bf16 v16bf;
typedef __attribute__((ext_vector_type(8)))  float  v8f;

#define BATCH     128
#define GRIDN     64
#define NPIX      4096      // 64*64
#define EMB       64
#define HDIM      128
#define ODIM      128
#define TILE_PIX  128
#define TILES     32        // NPIX / TILE_PIX

// bf16 table offsets (in bf16 elements) inside workspace
#define CE_OFF    0         // 16*64   = 1024
#define ROW_OFF   1024      // 64*32   = 2048
#define COL_OFF   3072      // 64*32   = 2048
#define W1_OFF    5120      // 128*128 = 16384  (B-fragment packed)
#define W2_OFF    21504     // 128*64  = 8192   (packed)
#define KW_OFF    29696     // 64*64   = 4096   (packed; vw contiguous after)
#define VW_OFF    33792     // 64*64   = 4096   (packed)
#define OW1_OFF   37888     // 128*128 = 16384  (packed)
#define OW2_OFF   54272     // 128*64  = 8192   (packed)
#define BF_TOTAL  62464

#define F32_BYTE_OFF 131072     // f32 region starts at 128 KB (alignment)
// f32 offsets (in floats) inside f32 region
#define CUR_OFF   0                         // B*64
#define Q_OFF     8192                      // B*64
#define TP_OFF    16384                     // B*TILES*66  (m, l, acc[64])
#define OS_OFF    (16384 + BATCH*TILES*66)  // B*TILES*64  (column sums)

// ---------------------------------------------------------------------------
// WMMA fragment helpers (layouts per CDNA5 ISA 7.12.2, wave32)
// ---------------------------------------------------------------------------
__device__ __forceinline__ v8f vzero8() {
  v8f z;
#pragma unroll
  for (int i = 0; i < 8; ++i) z[i] = 0.0f;
  return z;
}

// A fragment: 16x32 bf16, src is row-major [16][ld], lane&15 = row,
// per-lane K slots: k = ((j>>3)<<4) + (lane>>4)*8 + (j&7)
// -> two contiguous 16B chunks per lane (vectorizes to 2x ds_load_b128)
__device__ __forceinline__ v16bf load_a_frag(const __bf16* src, int ld, int lane) {
  const int row  = lane & 15;
  const int half = lane >> 4;
  const __bf16* base = src + row * ld + half * 8;
  v16bf a;
#pragma unroll
  for (int j = 0; j < 16; ++j) {
    int k = ((j >> 3) << 4) + (j & 7);
    a[j] = base[k];
  }
  return a;
}

// B fragment from pre-packed weights: lane's 16 bf16 are contiguous (32B).
__device__ __forceinline__ v16bf load_b_packed(const __bf16* base) {
  return *(const v16bf*)base;          // 2x ds_load_b128
}

__device__ __forceinline__ v8f wmma_bf16(v16bf a, v16bf b, v8f c) {
  // v_wmma_f32_16x16x32_bf16  D = A*B + C
  return __builtin_amdgcn_wmma_f32_16x16x32_bf16(
      /*neg_a=*/false, a, /*neg_b=*/false, b,
      /*c_mod=*/(short)0, c, /*reuse_a=*/false, /*reuse_b=*/false);
}

// cooperative 16KB (1024 x uint4) global->LDS weight stage
__device__ __forceinline__ void stage16k(__bf16* dst, const __bf16* src) {
  const uint4* s = (const uint4*)src;
  uint4*       d = (uint4*)dst;
  for (int i = threadIdx.x; i < 1024; i += 256) d[i] = s[i];
}

// Build a 16x128 bf16 feature strip: [color_embed(64) | row(32) | col(32)]
// Each lane moves 4 contiguous bf16 (8B) per row; segments are 32-aligned so
// a 4-element chunk never straddles table boundaries.
__device__ __forceinline__ void build_feat(__bf16* strip, const int* grid,
                                           const __bf16* bft, int p0, int lane) {
  const int kk = lane * 4;
  for (int r = 0; r < 16; ++r) {
    const int p     = p0 + r;
    const int color = grid[p];
    const int rr    = p >> 6;
    const int cc    = p & 63;
    const __bf16* src;
    if (kk < 64)       src = bft + CE_OFF  + color * 64 + kk;
    else if (kk < 96)  src = bft + ROW_OFF + rr * 32 + (kk - 64);
    else               src = bft + COL_OFF + cc * 32 + (kk - 96);
    *(uint2*)(strip + r * 128 + kk) = *(const uint2*)src;
  }
}

// GEMM1: strip(16x128 feat) x W(128x128) + b, relu -> strip (in place, bf16).
// Packed weights streamed through 16KB LDS stage in two 64-K-row halves
// (each half = tiles [kc2=0..1][nc=0..7] = 16KB contiguous).
// All 8 B fragments of a K-chunk are loaded into registers before the WMMA
// chain so the DS loads overlap WMMA issue (progressive partial dscnt waits).
__device__ __forceinline__ void gemm_mlp128(__bf16* wbuf, __bf16* strip,
                                            const __bf16* wg, const float* bias,
                                            int lane) {
  v8f acc[8];
#pragma unroll
  for (int nc = 0; nc < 8; ++nc) acc[nc] = vzero8();

  for (int kh = 0; kh < 2; ++kh) {
    __syncthreads();
    stage16k(wbuf, wg + kh * 8192);       // packed half
    __syncthreads();
    const v16bf a0 = load_a_frag(strip + kh * 64,      128, lane);
    const v16bf a1 = load_a_frag(strip + kh * 64 + 32, 128, lane);
#pragma unroll
    for (int kc = 0; kc < 2; ++kc) {
      v16bf bb[8];
#pragma unroll
      for (int nc = 0; nc < 8; ++nc)
        bb[nc] = load_b_packed(wbuf + (kc * 8 + nc) * 512 + lane * 16);
      const v16bf a = kc ? a1 : a0;
#pragma unroll
      for (int nc = 0; nc < 8; ++nc)
        acc[nc] = wmma_bf16(a, bb[nc], acc[nc]);
    }
  }
  const int col  = lane & 15;
  const int half = lane >> 4;
#pragma unroll
  for (int nc = 0; nc < 8; ++nc) {
    const float b = bias[nc * 16 + col];
#pragma unroll
    for (int r = 0; r < 8; ++r) {
      float v = acc[nc][r] + b;
      v = v > 0.0f ? v : 0.0f;                      // relu
      strip[(half * 8 + r) * 128 + nc * 16 + col] = (__bf16)v;
    }
  }
}

// GEMM2: strip(16x128) x W(128x64) + b -> out[4] (f32 accumulators, D-layout)
__device__ __forceinline__ void gemm_proj64(__bf16* wbuf, const __bf16* strip,
                                            const __bf16* wg, const float* bias,
                                            int lane, v8f out[4]) {
  __syncthreads();
  stage16k(wbuf, wg);                     // packed 128x64 = 16 KB
  __syncthreads();
#pragma unroll
  for (int nc = 0; nc < 4; ++nc) out[nc] = vzero8();
#pragma unroll
  for (int kc = 0; kc < 4; ++kc) {
    const v16bf a = load_a_frag(strip + kc * 32, 128, lane);
    v16bf bb[4];
#pragma unroll
    for (int nc = 0; nc < 4; ++nc)
      bb[nc] = load_b_packed(wbuf + (kc * 4 + nc) * 512 + lane * 16);
#pragma unroll
    for (int nc = 0; nc < 4; ++nc)
      out[nc] = wmma_bf16(a, bb[nc], out[nc]);
  }
  const int col = lane & 15;
#pragma unroll
  for (int nc = 0; nc < 4; ++nc) {
    const float b = bias[nc * 16 + col];
#pragma unroll
    for (int r = 0; r < 8; ++r) out[nc][r] += b;
  }
}

// ---------------------------------------------------------------------------
// Kernel P1: f32 -> bf16 tables (embeddings, row-major)
// ---------------------------------------------------------------------------
__global__ void cvt_bf16_kernel(const float* __restrict__ src,
                                __bf16* __restrict__ dst, int n) {
  int i = blockIdx.x * 256 + threadIdx.x;
  if (i < n) dst[i] = (__bf16)src[i];
}

// ---------------------------------------------------------------------------
// Kernel P2: f32 W[K][NT*16] -> bf16 WMMA-B-fragment-packed layout.
// dst[ti*512 + lane*16 + j] = W[kc*32 + (j>>3)*16 + (lane>>4)*8 + (j&7)][nc*16 + (lane&15)]
// with ti = kc*NT + nc.
// ---------------------------------------------------------------------------
__global__ void pack_b_kernel(const float* __restrict__ src,
                              __bf16* __restrict__ dst, int NT, int n) {
  int o = blockIdx.x * 256 + threadIdx.x;
  if (o >= n) return;
  const int ti   = o >> 9;           // / 512
  const int lane = (o >> 4) & 31;
  const int j    = o & 15;
  const int kc   = ti / NT;
  const int nc   = ti % NT;
  const int half = lane >> 4;
  const int col  = lane & 15;
  const int K = kc * 32 + (j >> 3) * 16 + half * 8 + (j & 7);
  const int N = nc * 16 + col;
  dst[o] = (__bf16)src[K * (NT * 16) + N];
}

// ---------------------------------------------------------------------------
// Kernel 0: per-batch first-nonzero mask scan -> cur, q
// ---------------------------------------------------------------------------
__global__ __launch_bounds__(256) void arc_curq_kernel(
    const int* __restrict__ obs, const float* __restrict__ rowe,
    const float* __restrict__ cole, const float* __restrict__ q_w,
    const float* __restrict__ q_b, float* __restrict__ wsf) {
  __shared__ int   minIdx;
  __shared__ float curS[64];
  const int b = blockIdx.x;
  if (threadIdx.x == 0) minIdx = NPIX;
  __syncthreads();
  const int* mask = obs + (size_t)b * 3 * NPIX + 2 * NPIX;
  int local = NPIX;
  for (int p = threadIdx.x; p < NPIX; p += 256)
    if (mask[p] != 0) { local = p; break; }        // strided -> first hit is min
  atomicMin(&minIdx, local);
  __syncthreads();
  const int  idx = minIdx;
  const bool has = idx < NPIX;
  const int  rr  = has ? (idx >> 6) : 0;
  const int  cc  = has ? (idx & 63) : 0;
  if (threadIdx.x < 64) {
    float v = 0.0f;
    if (has)
      v = (threadIdx.x < 32) ? rowe[rr * 32 + threadIdx.x]
                             : cole[cc * 32 + threadIdx.x - 32];
    curS[threadIdx.x] = v;
    wsf[CUR_OFF + b * 64 + threadIdx.x] = v;
  }
  __syncthreads();
  if (threadIdx.x < 64) {
    float acc = q_b[threadIdx.x];
    for (int j = 0; j < 64; ++j) acc += curS[j] * q_w[j * 64 + threadIdx.x];
    wsf[Q_OFF + b * 64 + threadIdx.x] = acc;
  }
}

// ---------------------------------------------------------------------------
// Kernel A: fused per-tile MLPs + k/v + online-softmax partial + output branch
// grid = B*TILES blocks, 256 threads (8 waves, each owns a 16-pixel strip)
// ---------------------------------------------------------------------------
__global__ __launch_bounds__(256) void arc_tile_kernel(
    const int* __restrict__ obs, const __bf16* __restrict__ bft,
    float* __restrict__ wsf, const float* __restrict__ in_b1,
    const float* __restrict__ in_b2, const float* __restrict__ k_b,
    const float* __restrict__ v_b, const float* __restrict__ out_b1,
    const float* __restrict__ out_b2) {
  __shared__ __bf16 wbuf[8192];        // 16 KB weight stage
  __shared__ __bf16 strips[8 * 2048];  // 8 waves x (16x128) bf16 = 32 KB
  __shared__ float  red[8 * 66];       // per-wave reduction scratch

  const int wave = threadIdx.x >> 5;
  const int lane = threadIdx.x & 31;
  const int b    = blockIdx.x / TILES;
  const int t    = blockIdx.x % TILES;
  const int p0   = t * TILE_PIX + wave * 16;
  __bf16* strip  = strips + wave * 2048;
  const int* igrid = obs + (size_t)b * 3 * NPIX;  // channel 0
  const int* ogrid = igrid + NPIX;                // channel 1

  __builtin_prefetch(bft + W1_OFF, 0, 1);         // global_prefetch_b8

  // ============ input branch: feat -> MLP -> e -> k,v -> scores ============
  build_feat(strip, igrid, bft, p0, lane);
  gemm_mlp128(wbuf, strip, bft + W1_OFF, in_b1, lane);   // h (relu) in strip

  v8f eacc[4];
  gemm_proj64(wbuf, strip, bft + W2_OFF, in_b2, lane, eacc);
  {
    const int col = lane & 15, half = lane >> 4;
#pragma unroll
    for (int nc = 0; nc < 4; ++nc)
#pragma unroll
      for (int r = 0; r < 8; ++r)
        strip[(half * 8 + r) * 64 + nc * 16 + col] = (__bf16)eacc[nc][r];
  }

  // k and v projections: stage kw|vw (contiguous packed, 16 KB)
  __syncthreads();
  stage16k(wbuf, bft + KW_OFF);
  __syncthreads();
  v8f kacc[4], vacc[4];
#pragma unroll
  for (int nc = 0; nc < 4; ++nc) { kacc[nc] = vzero8(); vacc[nc] = vzero8(); }
  {
    const v16bf a0 = load_a_frag(strip,      64, lane);
    const v16bf a1 = load_a_frag(strip + 32, 64, lane);
#pragma unroll
    for (int kc = 0; kc < 2; ++kc) {
      v16bf bk[4], bv[4];
#pragma unroll
      for (int nc = 0; nc < 4; ++nc) {
        bk[nc] = load_b_packed(wbuf +        (kc * 4 + nc) * 512 + lane * 16);
        bv[nc] = load_b_packed(wbuf + 4096 + (kc * 4 + nc) * 512 + lane * 16);
      }
      const v16bf a = kc ? a1 : a0;
#pragma unroll
      for (int nc = 0; nc < 4; ++nc) {
        kacc[nc] = wmma_bf16(a, bk[nc], kacc[nc]);
        vacc[nc] = wmma_bf16(a, bv[nc], vacc[nc]);
      }
    }
  }
  {
    const int col = lane & 15;
#pragma unroll
    for (int nc = 0; nc < 4; ++nc) {
      const float kb = k_b[nc * 16 + col], vb = v_b[nc * 16 + col];
#pragma unroll
      for (int r = 0; r < 8; ++r) { kacc[nc][r] += kb; vacc[nc][r] += vb; }
    }
  }

  // scores + online softmax partial over this wave's 16 pixels
  float qreg[4];
#pragma unroll
  for (int nc = 0; nc < 4; ++nc)
    qreg[nc] = wsf[Q_OFF + b * 64 + nc * 16 + (lane & 15)];

  float s[8];
#pragma unroll
  for (int r = 0; r < 8; ++r) {
    float p = 0.0f;
#pragma unroll
    for (int nc = 0; nc < 4; ++nc) p += qreg[nc] * kacc[nc][r];
#pragma unroll
    for (int m = 1; m < 16; m <<= 1) p += __shfl_xor(p, m, 32);
    s[r] = p * 0.125f;                               // / sqrt(E)
  }
  float mloc = s[0];
#pragma unroll
  for (int r = 1; r < 8; ++r) mloc = fmaxf(mloc, s[r]);
  const float mw = fmaxf(mloc, __shfl_xor(mloc, 16, 32));
  float w[8], lh = 0.0f;
#pragma unroll
  for (int r = 0; r < 8; ++r) { w[r] = __expf(s[r] - mw); lh += w[r]; }
  const float lw = lh + __shfl_xor(lh, 16, 32);
  float avn[4];
#pragma unroll
  for (int nc = 0; nc < 4; ++nc) {
    float a = 0.0f;
#pragma unroll
    for (int r = 0; r < 8; ++r) a += w[r] * vacc[nc][r];
    a += __shfl_xor(a, 16, 32);
    avn[nc] = a;
  }
  if (lane == 0) { red[wave * 66 + 0] = mw; red[wave * 66 + 1] = lw; }
  if (lane < 16) {
#pragma unroll
    for (int nc = 0; nc < 4; ++nc) red[wave * 66 + 2 + nc * 16 + lane] = avn[nc];
  }
  __syncthreads();
  if (threadIdx.x < 64) {
    float M = -3.0e38f;
    for (int ww = 0; ww < 8; ++ww) M = fmaxf(M, red[ww * 66]);
    float A = 0.0f, L = 0.0f;
    for (int ww = 0; ww < 8; ++ww) {
      const float e = __expf(red[ww * 66] - M);
      A += e * red[ww * 66 + 2 + threadIdx.x];
      L += e * red[ww * 66 + 1];
    }
    float* tp = wsf + TP_OFF + ((size_t)b * TILES + t) * 66;
    tp[2 + threadIdx.x] = A;
    if (threadIdx.x == 0) { tp[0] = M; tp[1] = L; }
  }
  __syncthreads();

  // ============ output branch: ofeat MLP -> per-tile column sums ============
  build_feat(strip, ogrid, bft, p0, lane);
  gemm_mlp128(wbuf, strip, bft + OW1_OFF, out_b1, lane);
  v8f oacc[4];
  gemm_proj64(wbuf, strip, bft + OW2_OFF, out_b2, lane, oacc);

  float osums[4];
#pragma unroll
  for (int nc = 0; nc < 4; ++nc) {
    float ss = 0.0f;
#pragma unroll
    for (int r = 0; r < 8; ++r) ss += oacc[nc][r];
    ss += __shfl_xor(ss, 16, 32);
    osums[nc] = ss;
  }
  if (lane < 16) {
#pragma unroll
    for (int nc = 0; nc < 4; ++nc) red[wave * 66 + nc * 16 + lane] = osums[nc];
  }
  __syncthreads();
  if (threadIdx.x < 64) {
    float ss = 0.0f;
    for (int ww = 0; ww < 8; ++ww) ss += red[ww * 66 + threadIdx.x];
    wsf[OS_OFF + ((size_t)b * TILES + t) * 64 + threadIdx.x] = ss;
  }
}

// ---------------------------------------------------------------------------
// Kernel F: per-batch softmax combine + o_w proj + mean + head MLP
// ---------------------------------------------------------------------------
__global__ __launch_bounds__(128) void arc_final_kernel(
    const float* __restrict__ wsf, const float* __restrict__ o_w,
    const float* __restrict__ o_b, const float* __restrict__ c_w1,
    const float* __restrict__ c_b1, const float* __restrict__ c_w2,
    const float* __restrict__ c_b2, float* __restrict__ out) {
  __shared__ float att0[64];
  __shared__ float comb[192];
  __shared__ float hbuf[128];
  const int b   = blockIdx.x;
  const int tid = threadIdx.x;
  const float* tp = wsf + TP_OFF + (size_t)b * TILES * 66;
  const float* os = wsf + OS_OFF + (size_t)b * TILES * 64;

  if (tid < 64) {
    float M = -3.0e38f;
    for (int t = 0; t < TILES; ++t) M = fmaxf(M, tp[t * 66]);
    float A = 0.0f, L = 0.0f;
    for (int t = 0; t < TILES; ++t) {
      const float e = __expf(tp[t * 66] - M);
      A += e * tp[t * 66 + 2 + tid];
      L += e * tp[t * 66 + 1];
    }
    att0[tid] = A / L;
  }
  __syncthreads();
  if (tid < 64) {
    float acc = o_b[tid];
    for (int c = 0; c < 64; ++c) acc += att0[c] * o_w[c * 64 + tid];
    comb[tid] = acc;                                  // attended
    float ss = 0.0f;
    for (int t = 0; t < TILES; ++t) ss += os[t * 64 + tid];
    comb[64 + tid] = ss * (1.0f / NPIX);              // output_context
    comb[128 + tid] = wsf[CUR_OFF + b * 64 + tid];    // cur
  }
  __syncthreads();
  float acc = c_b1[tid];
  for (int i = 0; i < 192; ++i) acc += comb[i] * c_w1[i * 128 + tid];
  hbuf[tid] = acc > 0.0f ? acc : 0.0f;
  __syncthreads();
  float o = c_b2[tid];
  for (int j = 0; j < 128; ++j) o += hbuf[j] * c_w2[j * 128 + tid];
  out[(size_t)b * ODIM + tid] = o;
}

// ---------------------------------------------------------------------------
extern "C" void kernel_launch(void* const* d_in, const int* in_sizes, int n_in,
                              void* d_out, int out_size, void* d_ws, size_t ws_size,
                              hipStream_t stream) {
  const int*   obs    = (const int*)d_in[0];
  const float* ce     = (const float*)d_in[1];
  const float* rowe   = (const float*)d_in[2];
  const float* cole   = (const float*)d_in[3];
  const float* in_w1  = (const float*)d_in[4];
  const float* in_b1  = (const float*)d_in[5];
  const float* in_w2  = (const float*)d_in[6];
  const float* in_b2  = (const float*)d_in[7];
  const float* q_w    = (const float*)d_in[8];
  const float* q_b    = (const float*)d_in[9];
  const float* k_w    = (const float*)d_in[10];
  const float* k_b    = (const float*)d_in[11];
  const float* v_w    = (const float*)d_in[12];
  const float* v_b    = (const float*)d_in[13];
  const float* o_w    = (const float*)d_in[14];
  const float* o_b    = (const float*)d_in[15];
  const float* out_w1 = (const float*)d_in[16];
  const float* out_b1 = (const float*)d_in[17];
  const float* out_w2 = (const float*)d_in[18];
  const float* out_b2 = (const float*)d_in[19];
  const float* c_w1   = (const float*)d_in[20];
  const float* c_b1   = (const float*)d_in[21];
  const float* c_w2   = (const float*)d_in[22];
  const float* c_b2   = (const float*)d_in[23];

  __bf16* bft = (__bf16*)d_ws;
  float*  wsf = (float*)((char*)d_ws + F32_BYTE_OFF);

  // bf16 tables: embeddings stay row-major, weights go to B-fragment layout
  auto cvt = [&](const float* s, int off, int n) {
    cvt_bf16_kernel<<<(n + 255) / 256, 256, 0, stream>>>(s, bft + off, n);
  };
  auto packb = [&](const float* s, int off, int NT, int n) {
    pack_b_kernel<<<(n + 255) / 256, 256, 0, stream>>>(s, bft + off, NT, n);
  };
  cvt(ce,   CE_OFF,  16 * 64);
  cvt(rowe, ROW_OFF, 64 * 32);
  cvt(cole, COL_OFF, 64 * 32);
  packb(in_w1,  W1_OFF,  8, 128 * 128);
  packb(in_w2,  W2_OFF,  4, 128 * 64);
  packb(k_w,    KW_OFF,  4, 64 * 64);
  packb(v_w,    VW_OFF,  4, 64 * 64);
  packb(out_w1, OW1_OFF, 8, 128 * 128);
  packb(out_w2, OW2_OFF, 4, 128 * 64);

  arc_curq_kernel<<<BATCH, 256, 0, stream>>>(obs, rowe, cole, q_w, q_b, wsf);

  arc_tile_kernel<<<BATCH * TILES, 256, 0, stream>>>(
      obs, bft, wsf, in_b1, in_b2, k_b, v_b, out_b1, out_b2);

  arc_final_kernel<<<BATCH, 128, 0, stream>>>(
      wsf, o_w, o_b, c_w1, c_b1, c_w2, c_b2, (float*)d_out);
}
